// A_MSA_67001489817932
// MI455X (gfx1250) — matrix operand order repaired and checked by reading, not compiled, not run on hardware
//
#include <hip/hip_runtime.h>

// ---------------------------------------------------------------------------
// Problem constants (fixed shapes from the reference)
// ---------------------------------------------------------------------------
static constexpr int B_    = 4;
static constexpr int C_    = 96;
static constexpr int H_    = 256;
static constexpr int W_    = 256;
static constexpr int HEADS_= 8;
static constexpr int W2_   = 32;           // W / HEADS
static constexpr int HW_   = H_ * W_;      // 65536
static constexpr int C3_   = 3 * C_;       // 288
static constexpr int D_    = C_ * H_;      // 24576  (attention row length)
static constexpr float EPS_ = 1e-12f;

typedef __bf16 bf16;
typedef __attribute__((ext_vector_type(16))) __bf16 v16bf;
typedef __attribute__((ext_vector_type(8)))  float  v8f;

// ---------------------------------------------------------------------------
// CDNA5 WMMA bf16 fragment index helpers (wave32)
//   A 16x32 (MxK), 16-bit: lane l -> m = l&15, K-half = l>>4.
//     element e (0..15): vgpr v=e>>1 ; k = 16*(v>>2) + 8*(l>>4) + 2*(v&3) + (e&1)
//   B 32x16 (KxN), 16-bit: lane l -> n = l&15; element e -> k = a_k_of(l, e), same as A
//   C/D 16x16 f32: vgpr r, lane l -> m = r + 8*(l>>4), n = l&15
// ---------------------------------------------------------------------------
__device__ __forceinline__ int a_k_of(int lane, int e) {
  int v = e >> 1;
  return ((v >> 2) << 4) + ((lane >> 4) << 3) + ((v & 3) << 1) + (e & 1);
}

__device__ __forceinline__ v8f wmma_bf16(const v16bf& a, const v16bf& b, const v8f& c) {
  return __builtin_amdgcn_wmma_f32_16x16x32_bf16(false, a, false, b, (short)0, c,
                                                 false, false);
}

// ---------------------------------------------------------------------------
// K1: qkv conv1x1 as WMMA GEMM.  out[b,o,p] = sum_c w[o,c]*x[b,c,p] + bias[o]
// One wave per block; block computes a 16(o) x 64(p) tile, K=96 in 3 steps.
// Output stored bf16 (image layout [b][288][HW]).
// ---------------------------------------------------------------------------
__global__ void __launch_bounds__(32)
k_qkv_conv1x1(const float* __restrict__ x, const float* __restrict__ w,
              const float* __restrict__ bias, bf16* __restrict__ out) {
  const int lane = threadIdx.x;
  const int b  = blockIdx.z;
  const int o0 = blockIdx.y * 16;
  const int p0 = blockIdx.x * 64;
  const int kh = lane >> 4;
  const int m  = lane & 15;
  const int krow = (lane & 15) + (kh << 4);

  // A fragments: weights, 3 K-steps of 32 channels
  v16bf afrag[3];
  #pragma unroll
  for (int ks = 0; ks < 3; ++ks) {
    const float* wr = w + (size_t)(o0 + m) * C_ + ks * 32;
    #pragma unroll
    for (int e = 0; e < 16; ++e) afrag[ks][e] = (bf16)wr[a_k_of(lane, e)];
  }

  #pragma unroll
  for (int nt = 0; nt < 4; ++nt) {
    const int p = p0 + nt * 16;
    v8f acc;
    #pragma unroll
    for (int r = 0; r < 8; ++r) acc[r] = bias[o0 + r + (kh << 3)];
    #pragma unroll
    for (int ks = 0; ks < 3; ++ks) {
      const int c = ks * 32;
      const float* xr = x + ((size_t)b * C_ + c) * HW_ + p;
      v16bf bfrag;
      #pragma unroll
      for (int e = 0; e < 16; ++e) bfrag[e] = (bf16)xr[(size_t)a_k_of(lane, e) * HW_ + (lane & 15)];
      acc = wmma_bf16(afrag[ks], bfrag, acc);
    }
    const int n = lane & 15;
    #pragma unroll
    for (int r = 0; r < 8; ++r) {
      const int o = o0 + r + (kh << 3);
      out[((size_t)b * C3_ + o) * HW_ + p + n] = (bf16)acc[r];
    }
  }
}

// ---------------------------------------------------------------------------
// K2: depthwise 3x3 conv, zero padding.  bf16 in/out, fp32 math.
// ---------------------------------------------------------------------------
__global__ void __launch_bounds__(256)
k_dwconv3x3(const bf16* __restrict__ in, const float* __restrict__ w,
            const float* __restrict__ bias, bf16* __restrict__ out) {
  const size_t idx = (size_t)blockIdx.x * 256 + threadIdx.x;   // over B*C3*HW
  const int pw = (int)(idx & 255);
  const int ph = (int)((idx >> 8) & 255);
  const int bc = (int)(idx >> 16);                // b*C3 + ch
  const int ch = bc % C3_;
  const bf16* base = in + ((size_t)bc) * HW_;
  const float* wk = w + ch * 9;
  float acc = bias[ch];
  #pragma unroll
  for (int dy = -1; dy <= 1; ++dy) {
    const int hh = ph + dy;
    if (hh < 0 || hh >= H_) continue;
    #pragma unroll
    for (int dx = -1; dx <= 1; ++dx) {
      const int ww = pw + dx;
      if (ww < 0 || ww >= W_) continue;
      acc += wk[(dy + 1) * 3 + (dx + 1)] * (float)base[hh * W_ + ww];
    }
  }
  out[idx] = (bf16)acc;
}

// ---------------------------------------------------------------------------
// K3: repack image layout -> head-major rows.
// heads[(((sec*B+b)*HEADS+head)*32+i)*D + d] = img[b][sec*C+c][h][head*32+i],
// with d = c*H + h.
// ---------------------------------------------------------------------------
__global__ void __launch_bounds__(256)
k_repack(const bf16* __restrict__ img, bf16* __restrict__ heads) {
  const size_t idx = (size_t)blockIdx.x * 256 + threadIdx.x;   // over 3*B*HEADS*32*D
  const int d = (int)(idx % D_);
  size_t r = idx / D_;
  const int i = (int)(r % W2_);  r /= W2_;
  const int head = (int)(r % HEADS_); r /= HEADS_;
  const int b = (int)(r % B_);
  const int sec = (int)(r / B_);
  const int c = d >> 8;          // d / H
  const int h = d & 255;         // d % H
  heads[idx] = img[(((size_t)b * C3_ + sec * C_ + c) * H_ + h) * W_ + head * W2_ + i];
}

// ---------------------------------------------------------------------------
// K4: row L2 norms of q (rows 0..1023) and k (rows 1024..2047).
// ---------------------------------------------------------------------------
__global__ void __launch_bounds__(256)
k_row_norms(const bf16* __restrict__ heads, float* __restrict__ norms) {
  const int row = blockIdx.x;            // 0..2047
  const int sec = row >> 10;             // 0=q, 1=k
  const int r = row & 1023;              // (b*HEADS+head)*32 + i
  const bf16* p = heads + ((size_t)sec * B_ * HEADS_ * W2_ + r) * D_;
  float s = 0.f;
  for (int d = threadIdx.x; d < D_; d += 256) {
    const float v = (float)p[d];
    s += v * v;
  }
  __shared__ float red[256];
  red[threadIdx.x] = s;
  __syncthreads();
  for (int off = 128; off > 0; off >>= 1) {
    if ((int)threadIdx.x < off) red[threadIdx.x] += red[threadIdx.x + off];
    __syncthreads();
  }
  if (threadIdx.x == 0) norms[row] = sqrtf(red[0]);
}

// ---------------------------------------------------------------------------
// K5: S = q k^T (32x32 per (b,head)) via WMMA, K=24576 split over 8 waves;
// deterministic LDS serial sum over the 8 waves; fused norm/temperature scale + softmax.
// P stored fp32 [32 (b,head)][32][32].
// ---------------------------------------------------------------------------
__global__ void __launch_bounds__(256)
k_attn_softmax(const bf16* __restrict__ heads, const float* __restrict__ norms,
               const float* __restrict__ temperature, float* __restrict__ P) {
  const int bh = blockIdx.x;             // b*HEADS + head
  const int wave = threadIdx.x >> 5;
  const int lane = threadIdx.x & 31;
  __shared__ float Sp[8][32][32];        // 32 KB

  const bf16* qb = heads + (size_t)bh * W2_ * D_;
  const bf16* kb = heads + ((size_t)B_ * HEADS_ + bh) * W2_ * D_;

  const int kh = lane >> 4;
  const int m = lane & 15;
  const int krow = (lane & 15) + (kh << 4);

  v8f acc[2][2];
  #pragma unroll
  for (int mt = 0; mt < 2; ++mt)
    #pragma unroll
    for (int nt = 0; nt < 2; ++nt)
      #pragma unroll
      for (int r = 0; r < 8; ++r) acc[mt][nt][r] = 0.f;

  for (int step = 0; step < 96; ++step) {
    const int d0 = wave * 3072 + step * 32;
    v16bf aq[2], bk[2];
    #pragma unroll
    for (int mt = 0; mt < 2; ++mt) {
      const bf16* qr = qb + (size_t)(mt * 16 + m) * D_ + d0;
      #pragma unroll
      for (int e = 0; e < 16; ++e) aq[mt][e] = qr[a_k_of(lane, e)];
    }
    #pragma unroll
    for (int nt = 0; nt < 2; ++nt) {
      #pragma unroll
      for (int e = 0; e < 16; ++e)
        bk[nt][e] = kb[(size_t)(nt * 16 + m) * D_ + d0 + a_k_of(lane, e)];
    }
    #pragma unroll
    for (int mt = 0; mt < 2; ++mt)
      #pragma unroll
      for (int nt = 0; nt < 2; ++nt)
        acc[mt][nt] = wmma_bf16(aq[mt], bk[nt], acc[mt][nt]);
  }

  // deposit per-wave partials (disjoint slots -> no atomics needed)
  #pragma unroll
  for (int mt = 0; mt < 2; ++mt)
    #pragma unroll
    for (int nt = 0; nt < 2; ++nt)
      #pragma unroll
      for (int r = 0; r < 8; ++r)
        Sp[wave][mt * 16 + r + (kh << 3)][nt * 16 + (lane & 15)] = acc[mt][nt][r];
  __syncthreads();

  // deterministic reduction over waves into Sp[0]
  for (int t = threadIdx.x; t < 1024; t += 256) {
    const int i = t >> 5, j = t & 31;
    float s = 0.f;
    #pragma unroll
    for (int wv = 0; wv < 8; ++wv) s += Sp[wv][i][j];
    Sp[0][i][j] = s;
  }
  __syncthreads();

  if (threadIdx.x < 32) {
    const int i = threadIdx.x;
    const int head = bh & 7;
    const float temp = temperature[head];
    const float nq = fmaxf(norms[bh * 32 + i], EPS_);
    float row[32];
    float mx = -3.402823466e38f;
    #pragma unroll
    for (int j = 0; j < 32; ++j) {
      const float nk = fmaxf(norms[1024 + bh * 32 + j], EPS_);
      const float v = Sp[0][i][j] / (nq * nk) * temp;
      row[j] = v;
      mx = fmaxf(mx, v);
    }
    float sum = 0.f;
    #pragma unroll
    for (int j = 0; j < 32; ++j) { row[j] = expf(row[j] - mx); sum += row[j]; }
    const float inv = 1.f / sum;
    #pragma unroll
    for (int j = 0; j < 32; ++j)
      P[(size_t)bh * 1024 + i * 32 + j] = row[j] * inv;
  }
}

// ---------------------------------------------------------------------------
// K6: out = P @ v via WMMA (M=32 rows i, K=32 cols j, N tile of 64 d's),
// scattered back to image layout [b][c][h][head*32+i] as bf16.
// ---------------------------------------------------------------------------
__global__ void __launch_bounds__(32)
k_attn_apply(const float* __restrict__ P, const bf16* __restrict__ heads,
             bf16* __restrict__ out_img) {
  const int lane = threadIdx.x;
  const int bh = blockIdx.y;
  const int d0 = blockIdx.x * 64;
  const int b = bh >> 3, head = bh & 7;
  const int kh = lane >> 4;
  const int m = lane & 15;
  const int krow = (lane & 15) + (kh << 4);

  const float* Pb = P + (size_t)bh * 1024;
  const bf16* vb = heads + ((size_t)2 * B_ * HEADS_ + bh) * W2_ * D_;

  v16bf ap[2];
  #pragma unroll
  for (int mt = 0; mt < 2; ++mt) {
    const float* pr = Pb + (mt * 16 + m) * 32;
    #pragma unroll
    for (int e = 0; e < 16; ++e) ap[mt][e] = (bf16)pr[a_k_of(lane, e)];
  }

  #pragma unroll
  for (int nt = 0; nt < 4; ++nt) {
    const int d = d0 + nt * 16;
    v16bf bv;
    const bf16* vr = vb + d;
    #pragma unroll
    for (int e = 0; e < 16; ++e) bv[e] = vr[(size_t)a_k_of(lane, e) * D_ + (lane & 15)];
    #pragma unroll
    for (int mt = 0; mt < 2; ++mt) {
      v8f acc;
      #pragma unroll
      for (int r = 0; r < 8; ++r) acc[r] = 0.f;
      acc = wmma_bf16(ap[mt], bv, acc);
      const int dd = d + (lane & 15);
      const int c = dd >> 8, h = dd & 255;
      #pragma unroll
      for (int r = 0; r < 8; ++r) {
        const int i = mt * 16 + r + (kh << 3);
        out_img[(((size_t)b * C_ + c) * H_ + h) * W_ + head * W2_ + i] = (bf16)acc[r];
      }
    }
  }
}

// ---------------------------------------------------------------------------
// K7: proj conv1x1 (WMMA GEMM, M=96, K=96, N=pixels), fp32 output.
// ---------------------------------------------------------------------------
__global__ void __launch_bounds__(32)
k_proj_conv1x1(const bf16* __restrict__ img, const float* __restrict__ w,
               const float* __restrict__ bias, float* __restrict__ out) {
  const int lane = threadIdx.x;
  const int b  = blockIdx.z;
  const int o0 = blockIdx.y * 16;
  const int p0 = blockIdx.x * 64;
  const int kh = lane >> 4;
  const int m  = lane & 15;
  const int krow = (lane & 15) + (kh << 4);

  v16bf afrag[3];
  #pragma unroll
  for (int ks = 0; ks < 3; ++ks) {
    const float* wr = w + (size_t)(o0 + m) * C_ + ks * 32;
    #pragma unroll
    for (int e = 0; e < 16; ++e) afrag[ks][e] = (bf16)wr[a_k_of(lane, e)];
  }

  #pragma unroll
  for (int nt = 0; nt < 4; ++nt) {
    const int p = p0 + nt * 16;
    v8f acc;
    #pragma unroll
    for (int r = 0; r < 8; ++r) acc[r] = bias[o0 + r + (kh << 3)];
    #pragma unroll
    for (int ks = 0; ks < 3; ++ks) {
      const int c = ks * 32;
      const bf16* xr = img + ((size_t)b * C_ + c) * HW_ + p;
      v16bf bfrag;
      #pragma unroll
      for (int e = 0; e < 16; ++e) bfrag[e] = xr[(size_t)a_k_of(lane, e) * HW_ + (lane & 15)];
      acc = wmma_bf16(afrag[ks], bfrag, acc);
    }
    const int n = lane & 15;
    #pragma unroll
    for (int r = 0; r < 8; ++r) {
      const int o = o0 + r + (kh << 3);
      out[((size_t)b * C_ + o) * HW_ + p + n] = acc[r];
    }
  }
}

// ---------------------------------------------------------------------------
// Launch
// ---------------------------------------------------------------------------
extern "C" void kernel_launch(void* const* d_in, const int* in_sizes, int n_in,
                              void* d_out, int out_size, void* d_ws, size_t ws_size,
                              hipStream_t stream) {
  const float* x      = (const float*)d_in[0];
  const float* qkv_w  = (const float*)d_in[1];
  const float* qkv_b  = (const float*)d_in[2];
  const float* dw_w   = (const float*)d_in[3];
  const float* dw_b   = (const float*)d_in[4];
  const float* proj_w = (const float*)d_in[5];
  const float* proj_b = (const float*)d_in[6];
  const float* temp   = (const float*)d_in[7];
  float* out = (float*)d_out;

  // Workspace layout (bytes):
  //  bufA : bf16 [B][288][HW]  (conv1x1 out; reused for head-major q/k/v)
  //  bufB : bf16 [B][288][HW]  (dwconv out; first 50MB reused for attn image)
  //  norms: f32  [2048]
  //  P    : f32  [32][32][32]
  const size_t bufElems = (size_t)B_ * C3_ * HW_;   // 75,497,472
  char* ws = (char*)d_ws;
  bf16* bufA  = (bf16*)ws;
  bf16* bufB  = (bf16*)(ws + bufElems * 2);
  float* norms = (float*)(ws + bufElems * 4);
  float* Pm    = norms + 2048;

  // 1) qkv = conv1x1(x)
  k_qkv_conv1x1<<<dim3(HW_ / 64, C3_ / 16, B_), 32, 0, stream>>>(x, qkv_w, qkv_b, bufA);
  // 2) qkv = dwconv3x3(qkv)
  k_dwconv3x3<<<(int)(bufElems / 256), 256, 0, stream>>>(bufA, dw_w, dw_b, bufB);
  // 3) repack to head-major rows
  k_repack<<<(int)(bufElems / 256), 256, 0, stream>>>(bufB, bufA);
  // 4) row norms for q and k
  k_row_norms<<<2048, 256, 0, stream>>>(bufA, norms);
  // 5) S = q k^T, scale by 1/(|q||k|) * temperature, softmax -> P
  k_attn_softmax<<<B_ * HEADS_, 256, 0, stream>>>(bufA, norms, temp, Pm);
  // 6) out_img = P @ v (scatter back to image layout, bf16 into bufB)
  k_attn_apply<<<dim3(D_ / 64, B_ * HEADS_), 32, 0, stream>>>(Pm, bufA, bufB);
  // 7) final conv1x1 projection -> fp32 d_out
  k_proj_conv1x1<<<dim3(HW_ / 64, C_ / 16, B_), 32, 0, stream>>>(bufB, proj_w, proj_b, out);
}
